// Speller_11647951307332
// MI455X (gfx1250) — matrix operand order repaired and checked by reading, not compiled, not run on hardware
//
#include <hip/hip_runtime.h>
#include <math.h>

// LAS decoder for MI455X (gfx1250, wave32, WMMA 16x16x32 f16).
// Strategy:
//  - One-time: pack LSTM weights ([Wih|Whh] K-concat) and [Wh|Wv] (N-concat)
//    into WMMA B-fragment layout (f16, 32B contiguous per lane per tile).
//  - One-time: keyp/valp GEMM (32768x256x512) with WMMA, A staged via a
//    double-buffered LDS slab (one barrier per K-step).
//    keyp/valp (32MB fp32) stay resident in the 192MB L2 across all steps.
//  - Per step (x200): pack A fragments for LSTM1/LSTM2, WMMA gate GEMMs with
//    all 4 gates per wave (in-register elementwise, no LDS). B fragments for
//    each K-step are preloaded into a register array so the 8 WMMAs issue
//    back-to-back while later loads are still in flight.
//  - Attention: one wave per batch; VALU matvecs against L2-resident k/v.
// Workspace requirement: ~43 MB.

#define Bsz 64
#define Lsz 512
#define Tn  200
#define Cn  34

typedef __attribute__((ext_vector_type(16))) _Float16 v16h;
typedef __attribute__((ext_vector_type(8)))  float    v8f;

__device__ __forceinline__ float sigf(float x) { return 1.0f / (1.0f + __expf(-x)); }

// ---------------------------------------------------------------------------
// B-fragment packing: W is row-major [N][K]; B[k][n] = W[n][k].
// Tile (kt,nt) of 32(K)x16(N); lane l holds column n = nt*16 + (l&15),
// K = kt*32 + (j<8 ? 8*(l>>4) + j : 16 + 8*(l>>4) + (j-8))  (j = element 0..15 of v16h).
// Storage: tile index = nt*ntk + kt; flat half index = tile*512 + lane*16 + j.
// mode 0: K-concat of (s0: K<split, s1: K>=split)
// mode 1: N-concat of (s0: n<split,  s1: n>=split)
// ---------------------------------------------------------------------------
__global__ void pack_b_kernel(const float* __restrict__ s0, const float* __restrict__ s1,
                              int split, int mode, int Ntot, int Ktot,
                              _Float16* __restrict__ dst)
{
    int ntk = Ktot >> 5;
    int total = (Ntot >> 4) * ntk * 512;
    for (int tid = blockIdx.x * blockDim.x + threadIdx.x; tid < total;
         tid += gridDim.x * blockDim.x) {
        int tile = tid >> 9;
        int lane = (tid >> 4) & 31;
        int j    = tid & 15;
        int nt = tile / ntk, kt = tile - nt * ntk;
        int n = nt * 16 + (lane & 15);
        int k = kt * 32 + ((j < 8) ? (8 * (lane >> 4) + j) : (16 + 8 * (lane >> 4) + (j - 8)));
        float v;
        if (mode == 0) {
            v = (k < split) ? s0[(size_t)n * split + k]
                            : s1[(size_t)n * (Ktot - split) + (k - split)];
        } else {
            v = (n < split) ? s0[(size_t)n * Ktot + k]
                            : s1[(size_t)(n - split) * Ktot + k];
        }
        dst[tid] = (_Float16)v;
    }
}

// A-fragment index map: tile (mt,kt), lane l, element j:
//   row = mt*16 + (l&15);  v=j>>1, p=j&1, h=l>>4
//   k_local = v<4 ? 2v+p+8h : 16+2(v-4)+p+8h
__device__ __forceinline__ void pack_a_idx(int tid, int ntk, int& row, int& k)
{
    int tile = tid >> 9;
    int lane = (tid >> 4) & 31;
    int j    = tid & 15;
    int mt = tile / ntk, kt = tile - mt * ntk;
    row = mt * 16 + (lane & 15);
    int v = j >> 1, p = j & 1, h = lane >> 4;
    int kl = (v < 4) ? (2 * v + p + 8 * h) : (16 + 2 * (v - 4) + p + 8 * h);
    k = kt * 32 + kl;
}

// A1 = [embed(256) | ctx(128) | h1(512)]  (K=896, ntk=28)
__global__ void pack_a1_kernel(const int* __restrict__ tgt, const float* __restrict__ emb,
                               const float* __restrict__ ctx, const float* __restrict__ h1,
                               int t, _Float16* __restrict__ dst)
{
    const int ntk = 28;
    int total = 4 * ntk * 512;
    for (int tid = blockIdx.x * blockDim.x + threadIdx.x; tid < total;
         tid += gridDim.x * blockDim.x) {
        int row, k; pack_a_idx(tid, ntk, row, k);
        float v;
        if (k < 256)      { int tok = tgt[row * 201 + t]; v = emb[(size_t)tok * 256 + k]; }
        else if (k < 384) v = ctx[row * 128 + (k - 256)];
        else              v = h1[row * 512 + (k - 384)];
        dst[tid] = (_Float16)v;
    }
}

// A2 = [h1(512) | h2(512)]  (K=1024, ntk=32)
__global__ void pack_a2_kernel(const float* __restrict__ h1, const float* __restrict__ h2,
                               _Float16* __restrict__ dst)
{
    const int ntk = 32;
    int total = 4 * ntk * 512;
    for (int tid = blockIdx.x * blockDim.x + threadIdx.x; tid < total;
         tid += gridDim.x * blockDim.x) {
        int row, k; pack_a_idx(tid, ntk, row, k);
        float v = (k < 512) ? h1[row * 512 + k] : h2[row * 512 + (k - 512)];
        dst[tid] = (_Float16)v;
    }
}

// ---------------------------------------------------------------------------
// keyp/valp GEMM: (B*L=32768) x (N=256: [keyp|valp]) x (K=512)
// Block: 64 rows, all 256 cols; 8 waves: mt = w&3, nsel = w>>2 (8 N-tiles each).
// A staged per K-tile through a double-buffered LDS slab (1 barrier/iter);
// B fragments preloaded into a register array so WMMAs issue back-to-back.
// ---------------------------------------------------------------------------
__global__ void __launch_bounds__(256) keyval_kernel(const float* __restrict__ listener,
    const _Float16* __restrict__ wkv, const float* __restrict__ bh,
    const float* __restrict__ bv, float* __restrict__ keyp, float* __restrict__ valp)
{
    __shared__ __attribute__((aligned(32))) _Float16 aslab[2][2048]; // 2 x (4 m-tiles x 512)
    int tid = threadIdx.x, lane = tid & 31, wv = tid >> 5;
    int mt = wv & 3, nsel = wv >> 2;
    int base = blockIdx.x * 64;

    // decompose this thread's pack element once (8 elements, stride 256)
    int prow[8], pkl[8], pe[8];
    #pragma unroll
    for (int i = 0; i < 8; ++i) {
        int e = tid + 256 * i;
        int emt = e >> 9, elane = (e >> 4) & 31, ej = e & 15;
        int vv = ej >> 1, p = ej & 1, h = elane >> 4;
        pe[i]   = e;
        prow[i] = emt * 16 + (elane & 15);
        pkl[i]  = (vv < 4) ? (2 * vv + p + 8 * h) : (16 + 2 * (vv - 4) + p + 8 * h);
    }

    // prologue: pack K-tile 0 into buffer 0
    #pragma unroll
    for (int i = 0; i < 8; ++i)
        aslab[0][pe[i]] = (_Float16)listener[(size_t)(base + prow[i]) * 512 + pkl[i]];
    __syncthreads();

    v8f acc[8] = {};
    for (int kt = 0; kt < 32; ++kt) {
        int cur = kt & 1;
        // preload this K-step's 8 B fragments (one big load clause)
        v16h bfr[8];
        #pragma unroll
        for (int i = 0; i < 8; ++i) {
            int ntg = nsel * 8 + i;
            bfr[i] = *(const v16h*)(wkv + ((size_t)ntg * 32 + kt) * 512 + lane * 16);
        }
        v16h a = *(const v16h*)(aslab[cur] + mt * 512 + lane * 16);
        // pack next K-tile into the other buffer while WMMAs run
        if (kt + 1 < 32) {
            #pragma unroll
            for (int i = 0; i < 8; ++i)
                aslab[1 - cur][pe[i]] =
                    (_Float16)listener[(size_t)(base + prow[i]) * 512 + (kt + 1) * 32 + pkl[i]];
        }
        #pragma unroll
        for (int i = 0; i < 8; ++i)
            acc[i] = __builtin_amdgcn_wmma_f32_16x16x32_f16(false, a, false, bfr[i],
                                                            (short)0, acc[i], false, false);
        __syncthreads();
    }
    #pragma unroll
    for (int i = 0; i < 8; ++i) {
        int n = (nsel * 8 + i) * 16 + (lane & 15);
        #pragma unroll
        for (int r = 0; r < 8; ++r) {
            int grow = base + mt * 16 + r + 8 * (lane >> 4);
            float v = acc[i][r] + ((n < 128) ? bh[n] : bv[n - 128]);
            if (n < 128) keyp[(size_t)grow * 128 + n] = v;
            else         valp[(size_t)grow * 128 + (n - 128)] = v;
        }
    }
}

// ---------------------------------------------------------------------------
// LSTM cell: gates(64 x 2048) = Apk(64 x K) @ Wpk(K x 2048), then elementwise.
// Grid: 8 blocks x 64 units. Wave w: mt = w&3, cp = w>>2; owns all 4 gates of
// two 16-unit columns -> i/f/g/o combined entirely in registers (no LDS).
// B fragments preloaded per K-step so the 8 WMMAs issue back-to-back.
// ---------------------------------------------------------------------------
__global__ void __launch_bounds__(256) lstm_kernel(const _Float16* __restrict__ apk,
    const _Float16* __restrict__ wpk, int ntk,
    const float* __restrict__ bih, const float* __restrict__ bhh,
    float* __restrict__ cst, float* __restrict__ hst)
{
    int tid = threadIdx.x, lane = tid & 31, wv = tid >> 5;
    int mt = wv & 3, cp = wv >> 2;
    int ub = blockIdx.x * 64;

    v8f acc[8] = {};
    for (int kt = 0; kt < ntk; ++kt) {
        v16h bfr[8];
        #pragma unroll
        for (int c = 0; c < 8; ++c) {
            int g = c >> 1, q = c & 1;
            int ng  = g * 512 + ub + (cp * 2 + q) * 16;
            int ntg = ng >> 4;
            bfr[c] = *(const v16h*)(wpk + ((size_t)ntg * ntk + kt) * 512 + lane * 16);
        }
        v16h a = *(const v16h*)(apk + ((size_t)mt * ntk + kt) * 512 + lane * 16);
        #pragma unroll
        for (int c = 0; c < 8; ++c)
            acc[c] = __builtin_amdgcn_wmma_f32_16x16x32_f16(false, a, false, bfr[c],
                                                            (short)0, acc[c], false, false);
    }
    #pragma unroll
    for (int q = 0; q < 2; ++q) {
        int u = ub + (cp * 2 + q) * 16 + (lane & 15);   // hidden unit 0..511
        float bi_ = bih[0 * 512 + u] + bhh[0 * 512 + u];
        float bf_ = bih[1 * 512 + u] + bhh[1 * 512 + u];
        float bg_ = bih[2 * 512 + u] + bhh[2 * 512 + u];
        float bo_ = bih[3 * 512 + u] + bhh[3 * 512 + u];
        #pragma unroll
        for (int r = 0; r < 8; ++r) {
            int row = mt * 16 + r + 8 * (lane >> 4);
            float iv = sigf(acc[0 * 2 + q][r] + bi_);
            float fv = sigf(acc[1 * 2 + q][r] + bf_);
            float gv = tanhf(acc[2 * 2 + q][r] + bg_);
            float ov = sigf(acc[3 * 2 + q][r] + bo_);
            float cn = fv * cst[row * 512 + u] + iv * gv;
            cst[row * 512 + u] = cn;
            hst[row * 512 + u] = ov * tanhf(cn);
        }
    }
}

// ---------------------------------------------------------------------------
// Attention + logits: one wave per batch (8 blocks x 8 waves).
// q = h2@Ws.T+bs; energy = q . keyp[b]; softmax*mask/renorm; ctx = a@valp[b];
// logits = [h2,ctx]@Wout.T+bout.  keyp/valp are L2-resident.
// ---------------------------------------------------------------------------
__global__ void __launch_bounds__(256) attend_kernel(
    const float* __restrict__ h2, const float* __restrict__ Ws, const float* __restrict__ bs,
    const float* __restrict__ keyp, const float* __restrict__ valp,
    const int* __restrict__ lens, float* __restrict__ ctx,
    const float* __restrict__ Wout, const float* __restrict__ bout,
    float* __restrict__ out_logits, float* __restrict__ out_attn,
    int t, int write_out)
{
    __shared__ float qbuf[8][128];
    __shared__ float abuf[8][512];
    int tid = threadIdx.x, lane = tid & 31, wv = tid >> 5;
    int b = blockIdx.x * 8 + wv;
    const float* hrow = h2 + (size_t)b * 512;

    for (int qi = lane; qi < 128; qi += 32) {
        float s = bs[qi];
        const float* wr = Ws + (size_t)qi * 512;
        for (int k = 0; k < 512; ++k) s += hrow[k] * wr[k];
        qbuf[wv][qi] = s;
    }
    __syncthreads();

    float e[16];
    float mmax = -3.4e38f;
    int len = lens[b];
    #pragma unroll 4
    for (int i = 0; i < 16; ++i) {
        int l = lane + 32 * i;
        const float* kr = keyp + ((size_t)b * 512 + l) * 128;
        float s = 0.f;
        for (int k = 0; k < 128; ++k) s += qbuf[wv][k] * kr[k];
        e[i] = s;
        mmax = fmaxf(mmax, s);
    }
    for (int m = 16; m >= 1; m >>= 1) mmax = fmaxf(mmax, __shfl_xor(mmax, m, 32));

    float sall = 0.f, smask = 0.f, w[16];
    #pragma unroll
    for (int i = 0; i < 16; ++i) {
        int l = lane + 32 * i;
        float ex = __expf(e[i] - mmax);
        sall += ex;
        float mk = (b == 0 || l < len) ? 1.0f : 0.0f;
        w[i] = ex * mk;
        smask += w[i];
    }
    for (int m = 16; m >= 1; m >>= 1) {
        sall  += __shfl_xor(sall, m, 32);
        smask += __shfl_xor(smask, m, 32);
    }
    float inv   = 1.0f / sall;
    float scale = inv / fmaxf(smask * inv, 1e-12f);
    #pragma unroll
    for (int i = 0; i < 16; ++i) {
        int l = lane + 32 * i;
        float a = w[i] * scale;
        abuf[wv][l] = a;
        if (write_out) out_attn[((size_t)b * Tn + t) * 512 + l] = a;
    }
    __syncthreads();

    for (int vi = lane; vi < 128; vi += 32) {
        float s = 0.f;
        for (int l = 0; l < 512; ++l) s += abuf[wv][l] * valp[((size_t)b * 512 + l) * 128 + vi];
        ctx[(size_t)b * 128 + vi] = s;
        qbuf[wv][vi] = s;                 // reuse qbuf as ctx scratch for logits
    }
    __syncthreads();

    if (write_out) {
        for (int c = lane; c < Cn; c += 32) {
            float s = bout[c];
            const float* wr = Wout + (size_t)c * 640;
            for (int k = 0; k < 512; ++k) s += hrow[k] * wr[k];
            for (int v = 0; v < 128; ++v) s += qbuf[wv][v] * wr[512 + v];
            out_logits[((size_t)b * Tn + t) * Cn + c] = s;
        }
    }
}

__global__ void init_state_kernel(const float* __restrict__ h10, const float* __restrict__ c10,
                                  const float* __restrict__ h20, const float* __restrict__ c20,
                                  float* __restrict__ h1, float* __restrict__ c1,
                                  float* __restrict__ h2, float* __restrict__ c2,
                                  float* __restrict__ ctx)
{
    int tid = blockIdx.x * blockDim.x + threadIdx.x;
    if (tid < 64 * 512) { h1[tid] = h10[tid]; c1[tid] = c10[tid];
                          h2[tid] = h20[tid]; c2[tid] = c20[tid]; }
    if (tid < 64 * 128) ctx[tid] = 0.0f;
}

extern "C" void kernel_launch(void* const* d_in, const int* in_sizes, int n_in,
                              void* d_out, int out_size, void* d_ws, size_t ws_size,
                              hipStream_t stream)
{
    const float* listener = (const float*)d_in[0];
    const int*   lens     = (const int*)d_in[1];
    const int*   tgt      = (const int*)d_in[2];
    /* d_in[3] teacher_forcing: reference ignores its value */
    const float* emb  = (const float*)d_in[4];
    const float* Ws   = (const float*)d_in[5];
    const float* bs   = (const float*)d_in[6];
    const float* Wh   = (const float*)d_in[7];
    const float* bh   = (const float*)d_in[8];
    const float* Wv   = (const float*)d_in[9];
    const float* bv   = (const float*)d_in[10];
    const float* Wih1 = (const float*)d_in[11];
    const float* bih1 = (const float*)d_in[12];
    const float* Whh1 = (const float*)d_in[13];
    const float* bhh1 = (const float*)d_in[14];
    const float* Wih2 = (const float*)d_in[15];
    const float* bih2 = (const float*)d_in[16];
    const float* Whh2 = (const float*)d_in[17];
    const float* bhh2 = (const float*)d_in[18];
    const float* Wout = (const float*)d_in[19];
    const float* bout = (const float*)d_in[20];
    const float* h10  = (const float*)d_in[21];
    const float* c10  = (const float*)d_in[22];
    const float* h20  = (const float*)d_in[23];
    const float* c20  = (const float*)d_in[24];

    char* ws = (char*)d_ws;
    size_t o = 0;
    float*    keyp = (float*)(ws + o);    o += (size_t)Bsz * Lsz * 128 * 4;  // 16 MB
    float*    valp = (float*)(ws + o);    o += (size_t)Bsz * Lsz * 128 * 4;  // 16 MB
    _Float16* wc1  = (_Float16*)(ws + o); o += (size_t)128 * 28 * 1024;      // 3.5 MB
    _Float16* wc2  = (_Float16*)(ws + o); o += (size_t)128 * 32 * 1024;      // 4 MB
    _Float16* wkv  = (_Float16*)(ws + o); o += (size_t)16 * 32 * 1024;       // 0.5 MB
    _Float16* a1   = (_Float16*)(ws + o); o += (size_t)4 * 28 * 1024;
    _Float16* a2   = (_Float16*)(ws + o); o += (size_t)4 * 32 * 1024;
    float*    h1   = (float*)(ws + o);    o += 64 * 512 * 4;
    float*    c1   = (float*)(ws + o);    o += 64 * 512 * 4;
    float*    h2   = (float*)(ws + o);    o += 64 * 512 * 4;
    float*    c2   = (float*)(ws + o);    o += 64 * 512 * 4;
    float*    ctx  = (float*)(ws + o);    o += 64 * 128 * 4;

    float* out_logits = (float*)d_out;
    float* out_attn   = out_logits + (size_t)Bsz * Tn * Cn;

    // One-time packs + projections
    pack_b_kernel<<<(128 * 28 * 512 + 255) / 256, 256, 0, stream>>>(Wih1, Whh1, 384, 0, 2048, 896, wc1);
    pack_b_kernel<<<(128 * 32 * 512 + 255) / 256, 256, 0, stream>>>(Wih2, Whh2, 512, 0, 2048, 1024, wc2);
    pack_b_kernel<<<(16 * 32 * 512 + 255) / 256, 256, 0, stream>>>(Wh, Wv, 128, 1, 256, 512, wkv);
    init_state_kernel<<<128, 256, 0, stream>>>(h10, c10, h20, c20, h1, c1, h2, c2, ctx);
    keyval_kernel<<<512, 256, 0, stream>>>(listener, wkv, bh, bv, keyp, valp);

    // Bootstrap: ctx0 = attend(h2_0), no outputs written
    attend_kernel<<<8, 256, 0, stream>>>(h2, Ws, bs, keyp, valp, lens, ctx,
                                         Wout, bout, out_logits, out_attn, 0, 0);

    for (int t = 0; t < Tn; ++t) {
        pack_a1_kernel<<<224, 256, 0, stream>>>(tgt, emb, ctx, h1, t, a1);
        lstm_kernel<<<8, 256, 0, stream>>>(a1, wc1, 28, bih1, bhh1, c1, h1);
        pack_a2_kernel<<<256, 256, 0, stream>>>(h1, h2, a2);
        lstm_kernel<<<8, 256, 0, stream>>>(a2, wc2, 32, bih2, bhh2, c2, h2);
        attend_kernel<<<8, 256, 0, stream>>>(h2, Ws, bs, keyp, valp, lens, ctx,
                                             Wout, bout, out_logits, out_attn, t, 1);
    }
}
